// SoftTree_56942676410675
// MI455X (gfx1250) — matrix operand order repaired and checked by reading, not compiled, not run on hardware
//
#include <hip/hip_runtime.h>

typedef __attribute__((ext_vector_type(16))) __bf16 v16bf;
typedef __attribute__((ext_vector_type(8)))  float  v8f;

#define N_NODES 1023
#define BM      32

// LDS strides (in elements) — padded so row offsets stagger across 64 banks
#define XSTR 521      // f32, 521 % 64 = 9 (coprime with 64)
#define PSTR 1033     // f32
#define QSTR 1042     // bf16 (even -> pair loads stay dword aligned)

// LDS byte offsets
#define X_OFF   0
#define P_OFF   (BM * XSTR * 4)            //  66688
#define PH_OFF  0                          //  reuses x region (dead after stage 1)
#define PL_OFF  (P_OFF + BM * PSTR * 4)    //  198912
#define O_OFF   (PL_OFF + BM * QSTR * 2)   //  265600
#define LDS_BYTES (O_OFF + BM * 64 * 4)    //  273792  (< 320 KB)

union Frag {
    v16bf    v;
    __bf16   h[16];
    unsigned u[8];
    uint4    q[2];
};

__device__ __forceinline__ void split_bf16(float a, __bf16& hi, __bf16& lo) {
    hi = (__bf16)a;
    lo = (__bf16)(a - (float)hi);
}

// ---------------------------------------------------------------------------
// Phase 0a: W (512 x 1023 f32) -> Wh/Wl bf16 in B-fragment order.
// Tile (nt in [0,64), kt in [0,16)) id = nt*16+kt; 512 elems per tile:
// lane L<16: K = kt*32 + e, L>=16: K = kt*32 + 16 + e; col = nt*16 + L%16.
// ---------------------------------------------------------------------------
__global__ __launch_bounds__(256) void conv_W(const float* __restrict__ W,
                                              __bf16* __restrict__ Wh,
                                              __bf16* __restrict__ Wl) {
    int tid  = blockIdx.x * 256 + threadIdx.x;   // < 524288
    int tile = tid >> 9;
    int r    = tid & 511;
    int lane = r >> 4;
    int e    = r & 15;
    int nt   = tile >> 4;
    int kt   = tile & 15;
    int k    = kt * 32 + ((lane < 16) ? e : 16 + e);
    int n    = nt * 16 + (lane & 15);
    float v  = (n < N_NODES) ? W[(size_t)k * N_NODES + n] : 0.0f;
    __bf16 hi, lo; split_bf16(v, hi, lo);
    Wh[tid] = hi; Wl[tid] = lo;
}

// Phase 0b: leaves (1024 x 64 f32) -> Lh/Ll, tiles (nt in [0,4), kt in [0,32))
__global__ __launch_bounds__(256) void conv_L(const float* __restrict__ L,
                                              __bf16* __restrict__ Lh,
                                              __bf16* __restrict__ Ll) {
    int tid  = blockIdx.x * 256 + threadIdx.x;   // < 65536
    int tile = tid >> 9;
    int r    = tid & 511;
    int lane = r >> 4;
    int e    = r & 15;
    int nt   = tile >> 5;
    int kt   = tile & 31;
    int k    = kt * 32 + ((lane < 16) ? e : 16 + e);
    int n    = nt * 16 + (lane & 15);
    float v  = L[(size_t)k * 64 + n];
    __bf16 hi, lo; split_bf16(v, hi, lo);
    Lh[tid] = hi; Ll[tid] = lo;
}

// ---------------------------------------------------------------------------
// Fused main kernel: logits -> sigmoid -> tree walk -> prob @ leaves
// ---------------------------------------------------------------------------
__global__ __launch_bounds__(256) void soft_tree_main(
        const float*  __restrict__ x,
        const float*  __restrict__ bias,
        const __bf16* __restrict__ Wh, const __bf16* __restrict__ Wl,
        const __bf16* __restrict__ Lh, const __bf16* __restrict__ Ll,
        float* __restrict__ out) {
    extern __shared__ char smem[];
    float*  x_lds = (float*)(smem + X_OFF);
    float*  p_lds = (float*)(smem + P_OFF);
    __bf16* ph    = (__bf16*)(smem + PH_OFF);
    __bf16* pl    = (__bf16*)(smem + PL_OFF);
    float*  o_lds = (float*)(smem + O_OFF);

    const int tid    = threadIdx.x;
    const int w      = tid >> 5;
    const int lane   = tid & 31;
    const int lrow   = lane & 15;
    const int kshift = (lane & 16) ? 16 : 0;
    const int row0   = blockIdx.x * BM;

    // ---- load x tile (32 x 512 f32) into padded LDS, coalesced ----
    for (int i = tid; i < BM * 512; i += 256) {
        int r = i >> 9, c = i & 511;
        x_lds[r * XSTR + c] = x[(size_t)(row0 + r) * 512 + c];
    }
    __syncthreads();

    // ================= Stage 1: logits = x @ W (split bf16) =================
    v8f acc[2][8];
#pragma unroll
    for (int a = 0; a < 2; ++a)
#pragma unroll
        for (int t = 0; t < 8; ++t) acc[a][t] = (v8f)0.0f;

    for (int kt = 0; kt < 16; ++kt) {
        Frag ah[2], al[2];
#pragma unroll
        for (int mt = 0; mt < 2; ++mt) {
            const float* src = x_lds + (mt * 16 + lrow) * XSTR;
            const int kb = kt * 32 + kshift;
#pragma unroll
            for (int j = 0; j < 8; ++j) {
                int k0 = kb + 2 * j;
                float a0 = src[k0], a1 = src[k0 + 1];
                __bf16 h0, l0, h1, l1;
                split_bf16(a0, h0, l0); split_bf16(a1, h1, l1);
                ah[mt].h[2 * j] = h0; ah[mt].h[2 * j + 1] = h1;
                al[mt].h[2 * j] = l0; al[mt].h[2 * j + 1] = l1;
            }
        }
#pragma unroll
        for (int t = 0; t < 8; ++t) {
            const size_t tb = (size_t)((w * 8 + t) * 16 + kt) * 512 + lane * 16;
            Frag bh, bl;
            bh.q[0] = ((const uint4*)(Wh + tb))[0];
            bh.q[1] = ((const uint4*)(Wh + tb))[1];
            bl.q[0] = ((const uint4*)(Wl + tb))[0];
            bl.q[1] = ((const uint4*)(Wl + tb))[1];
#pragma unroll
            for (int mt = 0; mt < 2; ++mt) {
                acc[mt][t] = __builtin_amdgcn_wmma_f32_16x16x32_bf16(
                    false, ah[mt].v, false, bh.v, (short)0, acc[mt][t], false, false);
                acc[mt][t] = __builtin_amdgcn_wmma_f32_16x16x32_bf16(
                    false, ah[mt].v, false, bl.v, (short)0, acc[mt][t], false, false);
                acc[mt][t] = __builtin_amdgcn_wmma_f32_16x16x32_bf16(
                    false, al[mt].v, false, bh.v, (short)0, acc[mt][t], false, false);
            }
        }
    }

    // ---- bias + sigmoid, p -> LDS ----
#pragma unroll
    for (int mt = 0; mt < 2; ++mt)
#pragma unroll
        for (int t = 0; t < 8; ++t) {
            int n = (w * 8 + t) * 16 + lrow;
            float bv = (n < N_NODES) ? bias[n] : 0.0f;
#pragma unroll
            for (int r = 0; r < 8; ++r) {
                int row = mt * 16 + r + ((lane & 16) ? 8 : 0);
                float z = acc[mt][t][r] + bv;
                float pv = __builtin_amdgcn_rcpf(1.0f + __expf(-z));
                p_lds[row * PSTR + n] = pv;
            }
        }
    __syncthreads();

    // ================= Stage 2: tree walk -> prob (bf16 hi/lo) =============
    {
        const int row = tid >> 3;     // 8 threads per row
        const int sub = tid & 7;      // top-3 leaf bits, constant per thread
        const float* prow = p_lds + row * PSTR;
        float pre = 1.0f;
#pragma unroll
        for (int lvl = 0; lvl < 3; ++lvl) {
            int idx = (1 << lvl) - 1 + (sub >> (3 - lvl));
            float v = prow[idx];
            pre *= ((sub >> (2 - lvl)) & 1) ? (1.0f - v) : v;
        }
        for (int li = 0; li < 128; ++li) {
            int l = sub * 128 + li;
            float prob = pre;
#pragma unroll
            for (int lvl = 3; lvl < 10; ++lvl) {
                int idx = (1 << lvl) - 1 + (l >> (10 - lvl));
                float v = prow[idx];
                prob *= ((l >> (9 - lvl)) & 1) ? (1.0f - v) : v;
            }
            __bf16 hi, lo; split_bf16(prob, hi, lo);
            ph[row * QSTR + l] = hi;
            pl[row * QSTR + l] = lo;
        }
    }
    // zero the output accumulator before the barrier
    for (int i = tid; i < BM * 64; i += 256) o_lds[i] = 0.0f;
    __syncthreads();

    // ================= Stage 3: out = prob @ leaves (split bf16) ============
    v8f oacc[2][4];
#pragma unroll
    for (int a = 0; a < 2; ++a)
#pragma unroll
        for (int t = 0; t < 4; ++t) oacc[a][t] = (v8f)0.0f;

#pragma unroll
    for (int kk = 0; kk < 4; ++kk) {
        const int kt = w * 4 + kk;     // 8 waves x 4 = 32 K-tiles
        Frag ah[2], al[2];
#pragma unroll
        for (int mt = 0; mt < 2; ++mt) {
            const int row = mt * 16 + lrow;
            const int kb  = kt * 32 + kshift;
#pragma unroll
            for (int j = 0; j < 8; ++j) {
                int k0 = kb + 2 * j;
                ah[mt].u[j] = *(const unsigned*)(ph + row * QSTR + k0);
                al[mt].u[j] = *(const unsigned*)(pl + row * QSTR + k0);
            }
        }
#pragma unroll
        for (int nt = 0; nt < 4; ++nt) {
            const size_t tb = (size_t)(nt * 32 + kt) * 512 + lane * 16;
            Frag bh, bl;
            bh.q[0] = ((const uint4*)(Lh + tb))[0];
            bh.q[1] = ((const uint4*)(Lh + tb))[1];
            bl.q[0] = ((const uint4*)(Ll + tb))[0];
            bl.q[1] = ((const uint4*)(Ll + tb))[1];
#pragma unroll
            for (int mt = 0; mt < 2; ++mt) {
                oacc[mt][nt] = __builtin_amdgcn_wmma_f32_16x16x32_bf16(
                    false, ah[mt].v, false, bh.v, (short)0, oacc[mt][nt], false, false);
                oacc[mt][nt] = __builtin_amdgcn_wmma_f32_16x16x32_bf16(
                    false, ah[mt].v, false, bl.v, (short)0, oacc[mt][nt], false, false);
                oacc[mt][nt] = __builtin_amdgcn_wmma_f32_16x16x32_bf16(
                    false, al[mt].v, false, bh.v, (short)0, oacc[mt][nt], false, false);
            }
        }
    }
    // cross-wave K-reduction via LDS float atomics (ds_add_f32)
#pragma unroll
    for (int mt = 0; mt < 2; ++mt)
#pragma unroll
        for (int nt = 0; nt < 4; ++nt) {
            int col = nt * 16 + lrow;
#pragma unroll
            for (int r = 0; r < 8; ++r) {
                int row = mt * 16 + r + ((lane & 16) ? 8 : 0);
                atomicAdd(&o_lds[row * 64 + col], oacc[mt][nt][r]);
            }
        }
    __syncthreads();

    // coalesced f32 store of the 32 x 64 tile
    for (int i = tid; i < BM * 64; i += 256) {
        int r = i >> 6, c = i & 63;
        out[(size_t)(row0 + r) * 64 + c] = o_lds[i];
    }
}

// ---------------------------------------------------------------------------
extern "C" void kernel_launch(void* const* d_in, const int* in_sizes, int n_in,
                              void* d_out, int out_size, void* d_ws, size_t ws_size,
                              hipStream_t stream) {
    const float* x      = (const float*)d_in[0];   // 8192 x 512
    const float* W      = (const float*)d_in[1];   // 512 x 1023
    const float* b      = (const float*)d_in[2];   // 1023
    const float* leaves = (const float*)d_in[3];   // 1024 x 64
    float* out = (float*)d_out;                    // 8192 x 64

    char* ws = (char*)d_ws;                        // needs 2.25 MB scratch
    __bf16* Wh = (__bf16*)ws;                      // 1 MB
    __bf16* Wl = Wh + (size_t)512 * 1024;          // 1 MB
    __bf16* Lh = (__bf16*)(ws + (size_t)2 * 1024 * 1024); // 128 KB
    __bf16* Ll = Lh + (size_t)1024 * 64;           // 128 KB

    conv_W<<<2048, 256, 0, stream>>>(W, Wh, Wl);
    conv_L<<<256,  256, 0, stream>>>(leaves, Lh, Ll);
    soft_tree_main<<<8192 / BM, 256, LDS_BYTES, stream>>>(x, b, Wh, Wl, Lh, Ll, out);
}
